// JointCodebookPredictor_40664750358545
// MI455X (gfx1250) — matrix operand order repaired and checked by reading, not compiled, not run on hardware
//
#include <hip/hip_runtime.h>
#include <hip/hip_bf16.h>

// ---------------------------------------------------------------------------
// JointCodebookPredictor for MI455X (gfx1250, wave32, WMMA bf16).
// - Main GEMM (16384x2048x6144) and logits GEMM on v_wmma_f32_16x16x32_bf16.
// - Global->LDS staging uses GLOBAL_LOAD_ASYNC_TO_LDS_B128 (ASYNCcnt) with
//   double-buffered LDS so the copy of tile i+1 overlaps WMMA on tile i.
// - linear_self transposed once -> one-hot self-regression becomes contiguous
//   streaming adds from L2 (192 MB L2 holds all weights).
// Workspace requirement: ~386 MB.
// d_out[0] = tot_logprob (f32), d_out[1] = tot_count (stored as f32).
// ---------------------------------------------------------------------------

typedef __attribute__((ext_vector_type(16))) __bf16 v16bf;
typedef __attribute__((ext_vector_type(8)))  float  v8f;
typedef __attribute__((ext_vector_type(4)))  int    v4i;

#define NCB 16
#define CS  256
#define HD  384
#define PD  2048
#define NF  16384
#define KH  (NCB * HD)        // 6144
#define LSR ((NCB - 1) * HD)  // 5760 rows of linear_self
#define LSC ((NCB - 1) * CS)  // 3840 cols of linear_self
#define BN_EPS 1e-5f

#define AS1 __attribute__((address_space(1)))
#define AS3 __attribute__((address_space(3)))

static __device__ __forceinline__ unsigned short f2bf(float f) {
  unsigned u = __float_as_uint(f);
  u += 0x7FFFu + ((u >> 16) & 1u);           // round-to-nearest-even
  return (unsigned short)(u >> 16);
}
static __device__ __forceinline__ float bf2f(unsigned short h) {
  return __uint_as_float(((unsigned)h) << 16);
}
static __device__ __forceinline__ __bf16 us2bf(unsigned short v) {
  union { unsigned short u; __bf16 b; } x; x.u = v; return x.b;
}

// 16-byte async copy global -> LDS (ASYNCcnt tracked).
static __device__ __forceinline__ void async_g2l_b128(void* lds, const void* g) {
#if defined(__HIP_DEVICE_COMPILE__) && __has_builtin(__builtin_amdgcn_global_load_async_to_lds_b128)
  __builtin_amdgcn_global_load_async_to_lds_b128(
      (AS1 v4i*)(unsigned long long)g,
      (AS3 v4i*)(unsigned)(unsigned long long)lds, 0, 0);
#elif defined(__HIP_DEVICE_COMPILE__)
  unsigned lo = (unsigned)(unsigned long long)lds;   // flat LDS addr low 32b = LDS offset
  asm volatile("global_load_async_to_lds_b128 %0, %1, off"
               :: "v"(lo), "v"(g) : "memory");
#else
  *reinterpret_cast<uint4*>(lds) = *reinterpret_cast<const uint4*>(g);  // host parse only
#endif
}
static __device__ __forceinline__ void wait_async0() {
#if defined(__HIP_DEVICE_COMPILE__) && __has_builtin(__builtin_amdgcn_s_wait_asynccnt)
  __builtin_amdgcn_s_wait_asynccnt(0);
#elif defined(__HIP_DEVICE_COMPILE__)
  asm volatile("s_wait_asynccnt 0x0" ::: "memory");
#endif
}

// ---------------------------------------------------------------- zero ------
__global__ void k_zero(float* __restrict__ out, float* __restrict__ sums) {
  int t = blockIdx.x * blockDim.x + threadIdx.x;
  if (t < 2) out[t] = 0.f;
  for (int i = t; i < 2 * KH; i += gridDim.x * blockDim.x) sums[i] = 0.f;
}

// ------------------------------------------------------- f32 -> bf16 -------
__global__ void k_cvt_bf16(const float* __restrict__ src,
                           unsigned short* __restrict__ dst, int n) {
  int i = blockIdx.x * blockDim.x + threadIdx.x;
  int stride = gridDim.x * blockDim.x;
  for (; i < n; i += stride) dst[i] = f2bf(src[i]);
}

// ------------------------------------------- transpose linear_self ---------
__global__ __launch_bounds__(256) void k_transpose(const float* __restrict__ in,
                                                   float* __restrict__ out) {
  __shared__ float tile[32][33];
  const int r0 = blockIdx.x * 32;   // LSR/32 = 180
  const int c0 = blockIdx.y * 32;   // LSC/32 = 120
  const int j  = threadIdx.x & 31;
  const int i0 = threadIdx.x >> 5;
#pragma unroll
  for (int s = 0; s < 4; ++s) {
    int i = i0 + s * 8;
    tile[i][j] = in[(size_t)(r0 + i) * LSC + (c0 + j)];
  }
  __syncthreads();
#pragma unroll
  for (int s = 0; s < 4; ++s) {
    int i = i0 + s * 8;
    out[(size_t)(c0 + i) * LSR + (r0 + j)] = tile[j][i];
  }
}

// ------------------------------------------------------ main GEMM ----------
// H[NF][KH] = A[NF][PD] @ W[KH][PD]^T  (bf16 in, f32 acc, bf16 out, no bias)
#define BM   128
#define BNB  128
#define BKK  64
#define LDT  72        // LDS row stride (bf16): 144 B, keeps b128 aligned
#define TILEB (BM * LDT * 2)        // 18432 B per operand tile
#define ARENA (4 * TILEB)           // 73728 B (>= 128*130*4 epilogue)

__global__ __launch_bounds__(256) void k_gemm1(
    const unsigned short* __restrict__ A, const unsigned short* __restrict__ W,
    unsigned short* __restrict__ H) {
  __shared__ __align__(16) char smem[ARENA];
  const int t    = threadIdx.x;
  const int wave = t >> 5;
  const int lane = t & 31;
  const int wm   = wave & 1;        // 2 waves along M
  const int wn   = wave >> 1;       // 4 waves along N
  const int m0   = blockIdx.x * BM;
  const int n0   = blockIdx.y * BNB;
  const int lm   = lane & 15;
  const int lh   = lane >> 4;

  v8f acc[4][2];
#pragma unroll
  for (int a = 0; a < 4; ++a)
#pragma unroll
    for (int b = 0; b < 2; ++b)
#pragma unroll
      for (int e = 0; e < 8; ++e) acc[a][b][e] = 0.f;

  auto tileA = [&](int buf) {
    return reinterpret_cast<unsigned short(*)[LDT]>(smem + buf * 2 * TILEB);
  };
  auto tileB = [&](int buf) {
    return reinterpret_cast<unsigned short(*)[LDT]>(smem + buf * 2 * TILEB + TILEB);
  };

  auto stage = [&](int buf, int k0) {       // async global -> LDS, 32 KB total
    unsigned short (*As)[LDT] = tileA(buf);
    unsigned short (*Bs)[LDT] = tileB(buf);
#pragma unroll
    for (int it = 0; it < 4; ++it) {
      int u = t + 256 * it;
      int row = u >> 3, q = u & 7;
      async_g2l_b128(&As[row][q * 8], A + (size_t)(m0 + row) * PD + k0 + q * 8);
      async_g2l_b128(&Bs[row][q * 8], W + (size_t)(n0 + row) * PD + k0 + q * 8);
    }
  };

  auto compute = [&](int buf) {
    unsigned short (*As)[LDT] = tileA(buf);
    unsigned short (*Bs)[LDT] = tileB(buf);
#pragma unroll
    for (int ks = 0; ks < 2; ++ks) {
      const int kb = ks * 32;
      v16bf af[4];
#pragma unroll
      for (int tm = 0; tm < 4; ++tm) {
        int row = wm * 64 + tm * 16 + lm;
#pragma unroll
        for (int e = 0; e < 8; ++e)  af[tm][e] = us2bf(As[row][kb + e + 8 * lh]);
#pragma unroll
        for (int e = 8; e < 16; ++e) af[tm][e] = us2bf(As[row][kb + e + 8 + 8 * lh]);
      }
      v16bf bf[2];
#pragma unroll
      for (int tn = 0; tn < 2; ++tn) {
        int col = wn * 32 + tn * 16 + lm;
#pragma unroll
        for (int e = 0; e < 16; ++e) bf[tn][e] = us2bf(Bs[col][kb + (e & 7) + 8 * lh + 16 * (e >> 3)]);
      }
#pragma unroll
      for (int tm = 0; tm < 4; ++tm)
#pragma unroll
        for (int tn = 0; tn < 2; ++tn)
          acc[tm][tn] = __builtin_amdgcn_wmma_f32_16x16x32_bf16(
              false, af[tm], false, bf[tn], (short)0, acc[tm][tn], false, false);
    }
  };

  const int nk = PD / BKK;                  // 32
  stage(0, 0);
  for (int i = 0; i < nk; ++i) {
    wait_async0();                          // my async stage landed in LDS
    __syncthreads();                        // everyone's stage landed
    if (i + 1 < nk) stage((i + 1) & 1, (i + 1) * BKK);  // overlap next copy
    compute(i & 1);
  }

  // Epilogue: accumulators -> LDS (f32, pad 130 avoids bank conflicts) ->
  // coalesced packed-bf16 stores.
  __syncthreads();
  float (*Csm)[130] = reinterpret_cast<float(*)[130]>(smem);
#pragma unroll
  for (int tm = 0; tm < 4; ++tm)
#pragma unroll
    for (int tn = 0; tn < 2; ++tn) {
      int col = wn * 32 + tn * 16 + lm;
#pragma unroll
      for (int r = 0; r < 8; ++r)
        Csm[wm * 64 + tm * 16 + r + 8 * lh][col] = acc[tm][tn][r];
    }
  __syncthreads();
#pragma unroll 4
  for (int it = 0; it < 32; ++it) {
    int lin = t + 256 * it;                 // 8192 packed pairs
    int row = lin >> 6, cp = lin & 63;
    unsigned u = (unsigned)f2bf(Csm[row][2 * cp]) |
                 ((unsigned)f2bf(Csm[row][2 * cp + 1]) << 16);
    *reinterpret_cast<unsigned*>(H + (size_t)(m0 + row) * KH + n0 + 2 * cp) = u;
  }
}

// ------------------------- self-regression gather + bias + relu (in place) -
__global__ __launch_bounds__(256) void k_selfpred(
    unsigned short* __restrict__ H, const float* __restrict__ b1,
    const float* __restrict__ selfT, const int* __restrict__ ci) {
  const int n = blockIdx.x;
  const int t = threadIdx.x;
  unsigned short* row = H + (size_t)n * KH;
  // pull the needed masked columns of linear_selfT toward this WGP
  for (int cb = 0; cb < NCB - 1; ++cb) {
    int cv = ci[n * NCB + cb];
    if (cv < 0) continue;
    int c = cb * CS + (cv > CS - 1 ? CS - 1 : cv);
    __builtin_prefetch(selfT + (size_t)c * LSR + cb * HD + t * 8, 0, 1);
  }
  float h[24];
#pragma unroll
  for (int i = 0; i < 24; ++i) {
    int ch = t + 256 * i;
    h[i] = bf2f(row[ch]) + b1[ch];
  }
  for (int cb = 0; cb < NCB - 1; ++cb) {
    int cv = ci[n * NCB + cb];
    if (cv < 0) continue;                       // invalid codebook index
    int c = cb * CS + (cv > CS - 1 ? CS - 1 : cv);
    const float* rp = selfT + (size_t)c * LSR;  // contiguous masked column
    int lo = (cb + 1) * HD;                     // mask: block k gets cb < k
#pragma unroll
    for (int i = 0; i < 24; ++i) {
      int ch = t + 256 * i;
      if (ch >= lo) h[i] += rp[ch - HD];
    }
  }
#pragma unroll
  for (int i = 0; i < 24; ++i) {
    int ch = t + 256 * i;
    float v = h[i] > 0.f ? h[i] : 0.f;          // relu
    row[ch] = f2bf(v);
  }
}

// ---------------------------------------------------- BN batch stats -------
__global__ __launch_bounds__(256) void k_stats(const unsigned short* __restrict__ H,
                                               float* __restrict__ sums) {
  int c  = blockIdx.x * 256 + threadIdx.x;
  int r0 = blockIdx.y * 256;
  float s = 0.f, s2 = 0.f;
  for (int r = 0; r < 256; ++r) {
    float v = bf2f(H[(size_t)(r0 + r) * KH + c]);
    s += v; s2 += v * v;
  }
  atomicAdd(&sums[c], s);
  atomicAdd(&sums[KH + c], s2);
}

__global__ void k_bnparam(const float* __restrict__ sums,
                          const float* __restrict__ gamma,
                          const float* __restrict__ beta,
                          float* __restrict__ bnp) {
  int c = blockIdx.x * 256 + threadIdx.x;
  float mean = sums[c] * (1.f / NF);
  float var  = sums[KH + c] * (1.f / NF) - mean * mean;  // biased variance
  float sc   = gamma[c] * rsqrtf(var + BN_EPS);
  bnp[c]      = sc;
  bnp[KH + c] = beta[c] - mean * sc;
}

// ----------------------- logits GEMM + log_softmax + loss ------------------
#define LDAH 392
#define LDB  40
__global__ __launch_bounds__(128) void k_logits(
    const unsigned short* __restrict__ H, const float* __restrict__ bnp,
    const unsigned short* __restrict__ L2, const float* __restrict__ bias2,
    const int* __restrict__ ci, float* __restrict__ out) {
  __shared__ __align__(16) unsigned short Ash[16][LDAH];
  __shared__ __align__(16) unsigned short Bsh[2][CS][LDB];   // double buffer
  __shared__ float Lsm[16][CS + 4];
  __shared__ float red[16][8];
  __shared__ float rmax[16];
  __shared__ float wsum, wcnt;

  const int t = threadIdx.x;
  const int wave = t >> 5, lane = t & 31, lm = lane & 15, lh = lane >> 4;
  const int f0 = blockIdx.x * 16;
  const int k  = blockIdx.y;

  if (t == 0) { wsum = 0.f; wcnt = 0.f; }

  auto stageB = [&](int buf, int kk) {          // async 256x32 bf16 slab
#pragma unroll
    for (int it = 0; it < 8; ++it) {
      int lin = t + 128 * it;                   // 1024 16-byte quads
      int c = lin >> 2, jq = lin & 3;
      async_g2l_b128(&Bsh[buf][c][jq * 8],
                     L2 + ((size_t)k * CS + c) * HD + kk * 32 + jq * 8);
    }
  };
  stageB(0, 0);

  // stage normalized hidden (BN applied on the fly) as bf16: 16 x 384
#pragma unroll 4
  for (int it = 0; it < 48; ++it) {
    int lin = t + 128 * it;
    int row = lin / HD, col = lin % HD;
    int ch = k * HD + col;
    float v = bf2f(H[(size_t)(f0 + row) * KH + ch]);
    v = v * bnp[ch] + bnp[KH + ch];
    Ash[row][col] = f2bf(v);
  }
  v8f acc[4];
#pragma unroll
  for (int q = 0; q < 4; ++q)
#pragma unroll
    for (int e = 0; e < 8; ++e) acc[q][e] = 0.f;

  for (int kk = 0; kk < HD / 32; ++kk) {
    wait_async0();
    __syncthreads();
    if (kk + 1 < HD / 32) stageB((kk + 1) & 1, kk + 1);
    v16bf af;
#pragma unroll
    for (int e = 0; e < 8; ++e)  af[e] = us2bf(Ash[lm][kk * 32 + e + 8 * lh]);
#pragma unroll
    for (int e = 8; e < 16; ++e) af[e] = us2bf(Ash[lm][kk * 32 + e + 8 + 8 * lh]);
#pragma unroll
    for (int q = 0; q < 4; ++q) {
      int colb = wave * 64 + q * 16;
      v16bf bfv;
#pragma unroll
      for (int e = 0; e < 16; ++e) bfv[e] = us2bf(Bsh[kk & 1][colb + lm][(e & 7) + 8 * lh + 16 * (e >> 3)]);
      acc[q] = __builtin_amdgcn_wmma_f32_16x16x32_bf16(
          false, af, false, bfv, (short)0, acc[q], false, false);
    }
  }
  __syncthreads();
#pragma unroll
  for (int q = 0; q < 4; ++q) {
    int col = wave * 64 + q * 16 + lm;
    float bb = bias2[k * CS + col];
#pragma unroll
    for (int r = 0; r < 8; ++r) Lsm[r + 8 * lh][col] = acc[q][r] + bb;
  }
  __syncthreads();
  // per-row (16 rows x 8 threads) log_softmax and one-hot pick
  const int row = t >> 3, s = t & 7;
  float pm = -3.0e38f;
  for (int j = 0; j < 32; ++j) { float v = Lsm[row][s * 32 + j]; pm = v > pm ? v : pm; }
  red[row][s] = pm;
  __syncthreads();
  if (s == 0) {
    float m = red[row][0];
    for (int j = 1; j < 8; ++j) m = red[row][j] > m ? red[row][j] : m;
    rmax[row] = m;
  }
  __syncthreads();
  float m = rmax[row];
  float ps = 0.f;
  for (int j = 0; j < 32; ++j) ps += __expf(Lsm[row][s * 32 + j] - m);
  red[row][s] = ps;
  __syncthreads();
  if (s == 0) {
    float l = 0.f;
    for (int j = 0; j < 8; ++j) l += red[row][j];
    int cv = ci[(f0 + row) * NCB + k];
    float valid = (cv >= 0) ? 1.f : 0.f;
    int tgt = cv < 0 ? 0 : (cv > CS - 1 ? CS - 1 : cv);
    float lp = (Lsm[row][tgt] - m - __logf(l)) * valid;
    atomicAdd(&wsum, lp);
    if (k == 0) atomicAdd(&wcnt, valid);
  }
  __syncthreads();
  if (t == 0) {
    atomicAdd(&out[0], wsum);
    if (k == 0) atomicAdd(&out[1], wcnt);
  }
}

// ---------------------------------------------------------------------------
extern "C" void kernel_launch(void* const* d_in, const int* in_sizes, int n_in,
                              void* d_out, int out_size, void* d_ws, size_t ws_size,
                              hipStream_t stream) {
  (void)in_sizes; (void)n_in; (void)out_size; (void)ws_size;
  const float* predictor = (const float*)d_in[0];
  const int*   ci        = (const int*)d_in[1];
  const float* W1        = (const float*)d_in[2];
  const float* b1        = (const float*)d_in[3];
  const float* lself     = (const float*)d_in[4];
  const float* gamma     = (const float*)d_in[5];
  const float* beta      = (const float*)d_in[6];
  const float* linear2   = (const float*)d_in[7];
  const float* bias2     = (const float*)d_in[8];
  float* out = (float*)d_out;

  char* ws = (char*)d_ws;
  size_t off = 0;
  auto alloc = [&](size_t bytes) -> char* {
    char* p = ws + off;
    off += (bytes + 255) & ~(size_t)255;
    return p;
  };
  unsigned short* A_bf  = (unsigned short*)alloc((size_t)NF * PD * 2);       //  64 MB
  unsigned short* W_bf  = (unsigned short*)alloc((size_t)KH * PD * 2);       //  24 MB
  unsigned short* L2_bf = (unsigned short*)alloc((size_t)NCB * CS * HD * 2); //   3 MB
  float*          selfT = (float*)alloc((size_t)LSC * LSR * 4);              //  88 MB
  unsigned short* H_bf  = (unsigned short*)alloc((size_t)NF * KH * 2);       // 192 MB
  float*          sums  = (float*)alloc((size_t)2 * KH * 4);
  float*          bnp   = (float*)alloc((size_t)2 * KH * 4);

  k_zero<<<48, 256, 0, stream>>>(out, sums);
  k_cvt_bf16<<<2048, 256, 0, stream>>>(predictor, A_bf, NF * PD);
  k_cvt_bf16<<<2048, 256, 0, stream>>>(W1, W_bf, KH * PD);
  k_cvt_bf16<<<512, 256, 0, stream>>>(linear2, L2_bf, NCB * CS * HD);
  k_transpose<<<dim3(LSR / 32, LSC / 32), 256, 0, stream>>>(lself, selfT);
  k_gemm1<<<dim3(NF / BM, KH / BNB), 256, 0, stream>>>(A_bf, W_bf, H_bf);
  k_selfpred<<<NF, 256, 0, stream>>>(H_bf, b1, selfT, ci);
  k_stats<<<dim3(KH / 256, NF / 256), 256, 0, stream>>>(H_bf, sums);
  k_bnparam<<<KH / 256, 256, 0, stream>>>(sums, gamma, beta, bnp);
  k_logits<<<dim3(NF / 16, NCB), 128, 0, stream>>>(H_bf, bnp, L2_bf, bias2, ci, out);
}
